// QKVMultiheadCrossAttention_25709674234097
// MI455X (gfx1250) — matrix operand order repaired and checked by reading, not compiled, not run on hardware
//
#include <hip/hip_runtime.h>

typedef _Float16 half_t;
typedef __attribute__((ext_vector_type(16))) _Float16 v16h;
typedef __attribute__((ext_vector_type(8)))  _Float16 v8h;
typedef __attribute__((ext_vector_type(2)))  _Float16 v2h;
typedef __attribute__((ext_vector_type(8)))  float    v8f;
typedef __attribute__((ext_vector_type(4)))  float    v4f;

#define HEADS   16
#define DHEAD   64
#define NCTX    2048
#define NKEY    2048
#define KT      64       // keys per iteration
#define NWAVES  8
#define QTILE   128      // NWAVES * 16 query rows per workgroup

// padded LDS strides (in halves) to avoid bank conflicts on fragment gathers
#define KSTR 66
#define VSTR 66
#define QSTR 72
#define PSTR 72

union U16 { v16h v; struct { v8h lo, hi; } h; };
union U2  { v16h v; v2h p[8]; };

// packed f32->f16 convert (v_cvt_pk_rtz_f16_f32); bit-cast __fp16x2 -> _Float16x2
__device__ __forceinline__ v2h pk_h2(float a, float b) {
  return __builtin_bit_cast(v2h, __builtin_amdgcn_cvt_pkrtz(a, b));
}

// one butterfly step of a 16-lane row-reduction, done on the VALU
__device__ __forceinline__ float pl16_max(float x, unsigned s0, unsigned s1) {
  unsigned xi = __builtin_bit_cast(unsigned, x);
  unsigned yi = __builtin_amdgcn_permlane16(xi, xi, s0, s1, false, false);
  return fmaxf(x, __builtin_bit_cast(float, yi));
}

__global__ __launch_bounds__(256)
void attn_fa_kernel(const float* __restrict__ qg,
                    const float* __restrict__ kvg,
                    float* __restrict__ outg)
{
  __shared__ alignas(16) half_t Kl[KT * KSTR];          // [key][c]   scaled f16
  __shared__ alignas(16) half_t Vl[DHEAD * VSTR];       // [c][key]   transposed f16
  __shared__ alignas(16) half_t Ql[QTILE * QSTR];       // [row][c]   scaled f16
  __shared__ alignas(16) half_t Pl[NWAVES * 16 * PSTR]; // per-wave P tiles

  const int tid  = threadIdx.x;
  const int wave = tid >> 5;
  const int lane = tid & 31;
  const int hi16 = lane >> 4;   // 0 or 1 (which 16-lane half)
  const int l16  = lane & 15;

  const int qblk = blockIdx.x;
  const int bh   = blockIdx.y;
  const int b    = bh >> 4;
  const int h    = bh & 15;

  // 64^(-1/4) on q and k; sqrt(log2 e) folded into both so S is in log2 units
  const float qk_scale = 0.35355339059327373f * 1.2011224087864498f;
  const int qbase = qblk * QTILE;

  // ---- stage Q tile (128x64) to LDS as scaled f16, coalesced b128 loads ----
  for (int i = tid; i < QTILE * (DHEAD / 4); i += 256) {
    int r = i >> 4, c4 = (i & 15) * 4;
    v4f v = *(const v4f*)(qg + (size_t)(b * NCTX + qbase + r) * 1024 + h * 64 + c4);
    *(v2h*)&Ql[r * QSTR + c4]     = pk_h2(v.x * qk_scale, v.y * qk_scale);
    *(v2h*)&Ql[r * QSTR + c4 + 2] = pk_h2(v.z * qk_scale, v.w * qk_scale);
  }
  __syncthreads();

  // ---- gather this wave's two 16x32 A-fragments of Q (contiguous k = 16*hi16 + s) ----
  const int ac = hi16 * 16;
  const half_t* qrow = &Ql[(wave * 16 + l16) * QSTR];
  U16 a0u, a1u;
  a0u.h.lo = *(const v8h*)&qrow[ac];
  a0u.h.hi = *(const v8h*)&qrow[ac + 8];
  a1u.h.lo = *(const v8h*)&qrow[32 + ac];
  a1u.h.hi = *(const v8h*)&qrow[32 + ac + 8];
  const v16h aq0 = a0u.v, aq1 = a1u.v;

  // constant all-ones B fragment: row-sums of P via the matrix pipe
  v16h vones;
#pragma unroll
  for (int j = 0; j < 16; ++j) vones[j] = (half_t)1.0f;

  // ---- flash-attention state: 4 output tiles + denominator tile ----
  v8f o0 = {}, o1 = {}, o2 = {}, o3 = {}, o4 = {};
  float mrow[8];
#pragma unroll
  for (int j = 0; j < 8; ++j) mrow[j] = -3.0e38f;

  const size_t kvbase = (size_t)(b * NKEY) * 2048 + (size_t)h * 128;
  half_t* Pw = &Pl[wave * 16 * PSTR];
  const int bko = hi16 * 16;   // B-layout: high half handles K+16

  for (int kb = 0; kb < NKEY; kb += KT) {
    __syncthreads();   // all waves done reading previous K/V tile
    // cooperative KV tile staging: b128 global loads, f32->f16, V transposed
#pragma unroll
    for (int k2 = 0; k2 < 4; ++k2) {
      int ii = tid + k2 * 256;
      int ss = ii >> 4, cc4 = (ii & 15) * 4;
      const float* rowp = kvg + kvbase + (size_t)(kb + ss) * 2048 + cc4;
      v4f kk = *(const v4f*)rowp;
      v4f vv = *(const v4f*)(rowp + 64);
      *(v2h*)&Kl[ss * KSTR + cc4]     = pk_h2(kk.x * qk_scale, kk.y * qk_scale);
      *(v2h*)&Kl[ss * KSTR + cc4 + 2] = pk_h2(kk.z * qk_scale, kk.w * qk_scale);
      Vl[(cc4 + 0) * VSTR + ss] = (half_t)vv.x;
      Vl[(cc4 + 1) * VSTR + ss] = (half_t)vv.y;
      Vl[(cc4 + 2) * VSTR + ss] = (half_t)vv.z;
      Vl[(cc4 + 3) * VSTR + ss] = (half_t)vv.w;
    }
    if (kb + KT < NKEY) {   // warm L2/L1 for the next tile -> global_prefetch_b8
      const float* nf = kvg + kvbase + (size_t)(kb + KT + (tid >> 2)) * 2048
                        + (size_t)(tid & 3) * 32;
      __builtin_prefetch(nf, 0, 0);
    }
    __syncthreads();   // tile visible to all waves

    // ---- S = Q K^T : 4 column tiles of 16 keys, K-dim 64 = 2 chained WMMA ----
    v8f st[4];
#pragma unroll
    for (int nt = 0; nt < 4; ++nt) {
      const half_t* kr = &Kl[(nt * 16 + l16) * KSTR];  // B: N = key = l16
      U2 b0, b1;
#pragma unroll
      for (int g = 0; g < 8; ++g) {                    // B: vgpr g holds K = 2g+bko, +1
        b0.p[g] = *(const v2h*)&kr[2 * g + bko];
        b1.p[g] = *(const v2h*)&kr[32 + 2 * g + bko];
      }
      v8f c = {};
      c = __builtin_amdgcn_wmma_f32_16x16x32_f16(false, aq0, false, b0.v, (short)0, c, false, false);
      c = __builtin_amdgcn_wmma_f32_16x16x32_f16(false, aq1, false, b1.v, (short)0, c, false, false);
      st[nt] = c;
    }

    // ---- online softmax on C-layout (row = j + 8*hi16, col = l16) ----
    // row max: permlane16 butterfly (xor 1,2,4,8) entirely on the VALU
#pragma unroll
    for (int j = 0; j < 8; ++j) {
      float m = fmaxf(fmaxf(st[0][j], st[1][j]), fmaxf(st[2][j], st[3][j]));
      m = pl16_max(m, 0x67452301u, 0xEFCDAB89u);   // xor 1
      m = pl16_max(m, 0x54761032u, 0xDCFE98BAu);   // xor 2
      m = pl16_max(m, 0x32107654u, 0xBA98FEDCu);   // xor 4
      m = pl16_max(m, 0xFEDCBA98u, 0x76543210u);   // xor 8
      float mn = fmaxf(mrow[j], m);
      float a = __builtin_amdgcn_exp2f(mrow[j] - mn);  // S already in log2 units
      mrow[j] = mn;
      o0[j] *= a; o1[j] *= a; o2[j] *= a; o3[j] *= a; o4[j] *= a;
    }
#pragma unroll
    for (int nt = 0; nt < 4; ++nt)
#pragma unroll
      for (int j = 0; j < 8; ++j)
        st[nt][j] = __builtin_amdgcn_exp2f(st[nt][j] - mrow[j]);  // raw v_exp_f32

    // ---- P (f16) -> per-wave LDS, C-layout scatter, row-major layout ----
    const int pmo = hi16 * 8;
#pragma unroll
    for (int nt = 0; nt < 4; ++nt)
#pragma unroll
      for (int j = 0; j < 8; ++j)
        Pw[(j + pmo) * PSTR + nt * 16 + l16] = (half_t)st[nt][j];

    asm volatile("s_wait_dscnt 0" ::: "memory");  // cross-lane LDS exchange within wave

    // ---- O += P V (+ ones column for row sums) ----
    const half_t* prow = &Pw[l16 * PSTR];
#pragma unroll
    for (int kc = 0; kc < 2; ++kc) {
      U16 pu;                                          // A-fragment of P
      pu.h.lo = *(const v8h*)&prow[kc * 32 + ac];
      pu.h.hi = *(const v8h*)&prow[kc * 32 + ac + 8];
      const v16h pa = pu.v;
#pragma unroll
      for (int ct = 0; ct < 4; ++ct) {
        const half_t* vr = &Vl[(ct * 16 + l16) * VSTR]; // B: N = out channel = l16
        U2 vb;
#pragma unroll
        for (int g = 0; g < 8; ++g)                     // K = key = kc*32 + 2g + bko
          vb.p[g] = *(const v2h*)&vr[kc * 32 + 2 * g + bko];
        v8f& acc = (ct == 0) ? o0 : (ct == 1) ? o1 : (ct == 2) ? o2 : o3;
        acc = __builtin_amdgcn_wmma_f32_16x16x32_f16(false, pa, false, vb.v, (short)0, acc, false, false);
      }
      // denominator: row-sum of P via all-ones B matrix
      o4 = __builtin_amdgcn_wmma_f32_16x16x32_f16(false, pa, false, vones, (short)0, o4, false, false);
    }
  }

  // ---- normalize by softmax denominator and store (coalesced f32) ----
  float* ob = outg + (size_t)(b * NCTX + qbase + wave * 16) * 1024 + h * 64;
  const int pmo = hi16 * 8;
#pragma unroll
  for (int j = 0; j < 8; ++j) {
    float inv = 1.0f / o4[j];
    float* orow = ob + (size_t)(j + pmo) * 1024;
    orow[ 0 + l16] = o0[j] * inv;
    orow[16 + l16] = o1[j] * inv;
    orow[32 + l16] = o2[j] * inv;
    orow[48 + l16] = o3[j] * inv;
  }
}

extern "C" void kernel_launch(void* const* d_in, const int* in_sizes, int n_in,
                              void* d_out, int out_size, void* d_ws, size_t ws_size,
                              hipStream_t stream) {
  (void)in_sizes; (void)n_in; (void)d_ws; (void)ws_size; (void)out_size;
  const float* q  = (const float*)d_in[0];
  const float* kv = (const float*)d_in[1];
  float* out = (float*)d_out;
  dim3 grid(NCTX / QTILE, 2 * HEADS);   // 16 q-tiles x (bs*heads)=32
  attn_fa_kernel<<<grid, 256, 0, stream>>>(q, kv, out);
}
